// FlaxMambaMixer_90417651515474
// MI455X (gfx1250) — compile-verified, matrix operand order checked by reading, not run on hardware
//
#include <hip/hip_runtime.h>
#include <hip/hip_bf16.h>

// ---------------------------------------------------------------------------
// Mamba mixer forward for MI455X (gfx1250), bf16 WMMA GEMMs + fused scan.
// B=1, L=2048, H=2048, D=4096, N=16, R=128, Kconv=4
// ---------------------------------------------------------------------------

#define L_SEQ 2048
#define H_DIM 2048
#define D_DIM 4096
#define N_ST  16
#define R_DT  128
#define NPAD  192          // R + 2N = 160 padded to multiple of 64

typedef __attribute__((ext_vector_type(16))) __bf16          v16bf;
typedef __attribute__((ext_vector_type(8)))  float           v8f;
typedef __attribute__((ext_vector_type(8)))  unsigned short  u16x8;
typedef __attribute__((ext_vector_type(16))) unsigned short  u16x16;

__device__ __forceinline__ unsigned short f32_to_bf16(float f) {
  unsigned int u = __float_as_uint(f);
  unsigned int r = u + 0x7FFFu + ((u >> 16) & 1u);   // round-to-nearest-even
  return (unsigned short)(r >> 16);
}

// ---------------------------------------------------------------------------
// elementwise f32 -> bf16
__global__ __launch_bounds__(256) void cvt_f32_bf16(const float* __restrict__ in,
                                                    unsigned short* __restrict__ out, int n) {
  int i = blockIdx.x * 256 + threadIdx.x;
  if (i < n) out[i] = f32_to_bf16(in[i]);
}

// sub-matrix f32 -> bf16 (strided read, packed write): out[r*cols+c]=in[r*ldin+c]
__global__ __launch_bounds__(256) void cvt_submat(const float* __restrict__ in,
                                                  unsigned short* __restrict__ out,
                                                  int rows, int cols, int ldin) {
  int i = blockIdx.x * 256 + threadIdx.x;
  if (i >= rows * cols) return;
  int r = i / cols, c = i - r * cols;
  out[i] = f32_to_bf16(in[(size_t)r * ldin + c]);
}

// tiled transpose + convert: in f32 [R,C] -> out bf16 [Cpad,R], pad rows zeroed
__global__ __launch_bounds__(256) void transpose_cvt(const float* __restrict__ in,
                                                     unsigned short* __restrict__ out,
                                                     int R, int C, int Cpad) {
  __shared__ float tile[32][33];
  int c0 = blockIdx.x * 32, r0 = blockIdx.y * 32;
  int tx = threadIdx.x;
  for (int j = threadIdx.y; j < 32; j += 8) {
    int r = r0 + j, c = c0 + tx;
    tile[j][tx] = (r < R && c < C) ? in[(size_t)r * C + c] : 0.0f;
  }
  __syncthreads();
  for (int j = threadIdx.y; j < 32; j += 8) {
    int c = c0 + j, r = r0 + tx;      // out[c][r]
    if (c < Cpad && r < R) out[(size_t)c * R + r] = f32_to_bf16(tile[tx][j]);
  }
}

// ---------------------------------------------------------------------------
// WMMA bf16 GEMM:  C[M,N] f32 = A[M,K] @ B, with B given transposed BT[N,K].
// Block = 256 threads = 8 waves (4 along M x 2 along N), wave tile 32x32,
// block tile 128x64.  All dims assumed multiples of the tile sizes.
__device__ __forceinline__ v16bf load_frag(const unsigned short* __restrict__ p) {
  u16x8 lo = *(const u16x8*)(p);
  u16x8 hi = *(const u16x8*)(p + 16);
  u16x16 v;
#pragma unroll
  for (int i = 0; i < 8; ++i) { v[i] = lo[i]; v[i + 8] = hi[i]; }
  return __builtin_bit_cast(v16bf, v);
}

__global__ __launch_bounds__(256) void gemm_bf16(const unsigned short* __restrict__ A,
                                                 const unsigned short* __restrict__ BT,
                                                 float* __restrict__ C,
                                                 int M, int N, int K) {
  int lane = threadIdx.x & 31;
  int wave = threadIdx.x >> 5;              // 0..7
  int wm = wave & 3, wn = wave >> 2;        // 4 x 2 wave grid
  int r0 = blockIdx.y * 128 + wm * 32;
  int c0 = blockIdx.x * 64 + wn * 32;
  int m  = lane & 15;
  int kh = (lane >> 4) << 3;                // lanes 16-31 take K-offset 8

  const unsigned short* Ar0 = A  + (size_t)(r0 + m) * K + kh;
  const unsigned short* Ar1 = A  + (size_t)(r0 + 16 + m) * K + kh;
  const unsigned short* Bc0 = BT + (size_t)(c0 + m) * K + kh;
  const unsigned short* Bc1 = BT + (size_t)(c0 + 16 + m) * K + kh;

  v8f acc00 = {}, acc01 = {}, acc10 = {}, acc11 = {};
  for (int k = 0; k < K; k += 32) {
    v16bf a0 = load_frag(Ar0 + k);
    v16bf a1 = load_frag(Ar1 + k);
    v16bf b0 = load_frag(Bc0 + k);
    v16bf b1 = load_frag(Bc1 + k);
    acc00 = __builtin_amdgcn_wmma_f32_16x16x32_bf16(false, a0, false, b0, (short)0, acc00, false, false);
    acc01 = __builtin_amdgcn_wmma_f32_16x16x32_bf16(false, a0, false, b1, (short)0, acc01, false, false);
    acc10 = __builtin_amdgcn_wmma_f32_16x16x32_bf16(false, a1, false, b0, (short)0, acc10, false, false);
    acc11 = __builtin_amdgcn_wmma_f32_16x16x32_bf16(false, a1, false, b1, (short)0, acc11, false, false);
  }
  int rowoff = (lane >> 4) << 3;            // C layout: VGPR j -> M=j (+8 for hi lanes)
  int col    = lane & 15;
#pragma unroll
  for (int j = 0; j < 8; ++j) {
    size_t ra = (size_t)(r0 + j + rowoff);
    size_t rb = (size_t)(r0 + 16 + j + rowoff);
    C[ra * N + c0 + col]      = acc00[j];
    C[ra * N + c0 + 16 + col] = acc01[j];
    C[rb * N + c0 + col]      = acc10[j];
    C[rb * N + c0 + 16 + col] = acc11[j];
  }
}

// ---------------------------------------------------------------------------
// causal depthwise conv (K=4) + SiLU.  hs_raw = proj[:, 0:D]
__global__ __launch_bounds__(256) void conv_silu(const float* __restrict__ proj,
                                                 const float* __restrict__ conv_w,
                                                 const float* __restrict__ conv_b,
                                                 float* __restrict__ hs,
                                                 unsigned short* __restrict__ hs_b) {
  int idx = blockIdx.x * 256 + threadIdx.x;      // t*D + d
  int t = idx >> 12, d = idx & (D_DIM - 1);
  float acc = conv_b[d];
#pragma unroll
  for (int k = 0; k < 4; ++k) {
    int tt = t - 3 + k;
    if (tt >= 0) acc += proj[(size_t)tt * (2 * D_DIM) + d] * conv_w[k * D_DIM + d];
  }
  float s = acc / (1.0f + __expf(-acc));         // SiLU
  hs[idx]   = s;
  hs_b[idx] = f32_to_bf16(s);
}

// dt = softplus(dt_raw + b_dt), in place
__global__ __launch_bounds__(256) void dt_softplus(float* __restrict__ dtb,
                                                   const float* __restrict__ b_dt) {
  int idx = blockIdx.x * 256 + threadIdx.x;
  float v = dtb[idx] + b_dt[idx & (D_DIM - 1)];
  dtb[idx] = (v > 20.0f) ? v : log1pf(__expf(v));
}

// ---------------------------------------------------------------------------
// fused selective scan + D-skip + output gating, writes y as bf16.
// One block handles 16 d-channels x 16 states; thread = (d_local, n).
#define SCAN_T 32
__global__ __launch_bounds__(256) void scan_kernel(const float* __restrict__ dt,
                                                   const float* __restrict__ hs,
                                                   const float* __restrict__ ssm_p,
                                                   const float* __restrict__ A_log,
                                                   const float* __restrict__ D_param,
                                                   const float* __restrict__ proj,
                                                   unsigned short* __restrict__ yb) {
  __shared__ float sh_dt[SCAN_T][16];
  __shared__ float sh_hs[SCAN_T][16];
  __shared__ float sh_B [SCAN_T][16];
  __shared__ float sh_C [SCAN_T][16];
  __shared__ float sh_g [SCAN_T][16];

  int tid = threadIdx.x;
  int dl = tid >> 4, n = tid & 15;
  int d0 = blockIdx.x * 16;
  int d  = d0 + dl;

  float Adn = -__expf(A_log[d * N_ST + n]);
  float Dp  = D_param[d];
  float s   = 0.0f;

  for (int t0 = 0; t0 < L_SEQ; t0 += SCAN_T) {
    __syncthreads();
    for (int i = tid; i < SCAN_T * 16; i += 256) {
      int tt = i >> 4, dd = i & 15;
      size_t trow = (size_t)(t0 + tt);
      sh_dt[tt][dd] = dt [trow * D_DIM + d0 + dd];
      sh_hs[tt][dd] = hs [trow * D_DIM + d0 + dd];
      sh_B [tt][dd] = ssm_p[trow * NPAD + R_DT + dd];
      sh_C [tt][dd] = ssm_p[trow * NPAD + R_DT + N_ST + dd];
      sh_g [tt][dd] = proj[trow * (2 * D_DIM) + D_DIM + d0 + dd];
    }
    __syncthreads();
#pragma unroll 4
    for (int tt = 0; tt < SCAN_T; ++tt) {
      float dtv = sh_dt[tt][dl];
      float hv  = sh_hs[tt][dl];
      float dA  = __expf(dtv * Adn);
      s = dA * s + dtv * sh_B[tt][n] * hv;
      float yv = s * sh_C[tt][n];
      yv += __shfl_xor(yv, 1, 32);             // reduce over n (16-lane groups)
      yv += __shfl_xor(yv, 2, 32);
      yv += __shfl_xor(yv, 4, 32);
      yv += __shfl_xor(yv, 8, 32);
      if (n == 0) {
        float g = sh_g[tt][dl];
        float o = (yv + hv * Dp) * (g / (1.0f + __expf(-g)));
        yb[(size_t)(t0 + tt) * D_DIM + d] = f32_to_bf16(o);
      }
    }
  }
}

// ---------------------------------------------------------------------------
extern "C" void kernel_launch(void* const* d_in, const int* in_sizes, int n_in,
                              void* d_out, int out_size, void* d_ws, size_t ws_size,
                              hipStream_t stream) {
  const float* x      = (const float*)d_in[0];   // [L,H]
  const float* W_in   = (const float*)d_in[1];   // [H,2D]
  const float* conv_w = (const float*)d_in[2];   // [4,D]
  const float* conv_b = (const float*)d_in[3];   // [D]
  const float* W_x    = (const float*)d_in[4];   // [D,160]
  const float* W_dt   = (const float*)d_in[5];   // [R,D]
  const float* b_dt   = (const float*)d_in[6];   // [D]
  const float* A_log  = (const float*)d_in[7];   // [D,N]
  const float* D_par  = (const float*)d_in[8];   // [D]
  const float* W_out  = (const float*)d_in[9];   // [D,H]
  float* out = (float*)d_out;                    // [L,H] f32
  (void)in_sizes; (void)n_in; (void)out_size; (void)ws_size;

  char* ws = (char*)d_ws;
  size_t off = 0;
  auto arena = [&](size_t b) { size_t o = off; off += (b + 255) & ~(size_t)255; return o; };

  float*          proj  = (float*)         (ws + arena((size_t)L_SEQ * 2 * D_DIM * 4)); // 64MB
  float*          hsf   = (float*)         (ws + arena((size_t)L_SEQ * D_DIM * 4));     // 32MB
  unsigned short* hsb   = (unsigned short*)(ws + arena((size_t)L_SEQ * D_DIM * 2));     // 16MB
  float*          ssm_p = (float*)         (ws + arena((size_t)L_SEQ * NPAD * 4));      // 1.5MB
  float*          dtb   = (float*)         (ws + arena((size_t)L_SEQ * D_DIM * 4));     // 32MB
  unsigned short* yb    = (unsigned short*)(ws + arena((size_t)L_SEQ * D_DIM * 2));     // 16MB
  char*           regA  =                   ws + arena((size_t)42 * 1024 * 1024);       // reused

  // region A, phase 1: xb + W_in^T          (phase 2 reuses the same space)
  unsigned short* xb    = (unsigned short*)(regA);
  unsigned short* winT  = (unsigned short*)(regA + ((size_t)L_SEQ * H_DIM * 2 + 255 & ~(size_t)255));
  // region A, phase 2 (after GEMM1): small weights
  unsigned short* wxT   = (unsigned short*)(regA);
  unsigned short* dtinb = (unsigned short*)(regA + 2 * 1024 * 1024);
  unsigned short* wdtT  = (unsigned short*)(regA + 3 * 1024 * 1024);
  unsigned short* woutT = (unsigned short*)(regA + 5 * 1024 * 1024);

  dim3 tb(32, 8);

  // 1) x -> bf16 ; W_in -> bf16 transposed [2D, H]
  cvt_f32_bf16<<<(L_SEQ * H_DIM) / 256, 256, 0, stream>>>(x, xb, L_SEQ * H_DIM);
  transpose_cvt<<<dim3((2 * D_DIM) / 32, H_DIM / 32), tb, 0, stream>>>(W_in, winT, H_DIM, 2 * D_DIM, 2 * D_DIM);

  // 2) proj[L,2D] = x @ W_in        (M=2048, N=8192, K=2048)
  gemm_bf16<<<dim3((2 * D_DIM) / 64, L_SEQ / 128), 256, 0, stream>>>(xb, winT, proj, L_SEQ, 2 * D_DIM, H_DIM);

  // 3) causal conv + SiLU -> hs (f32 + bf16)
  conv_silu<<<(L_SEQ * D_DIM) / 256, 256, 0, stream>>>(proj, conv_w, conv_b, hsf, hsb);

  // 4) ssm_p[L,192] = hs @ W_x      (W_x^T padded to 192 rows)
  transpose_cvt<<<dim3(NPAD / 32, D_DIM / 32), tb, 0, stream>>>(W_x, wxT, D_DIM, R_DT + 2 * N_ST, NPAD);
  gemm_bf16<<<dim3(NPAD / 64, L_SEQ / 128), 256, 0, stream>>>(hsb, wxT, ssm_p, L_SEQ, NPAD, D_DIM);

  // 5) dt_raw[L,D] = dt_in @ W_dt   (M=2048, N=4096, K=128)
  cvt_submat<<<(L_SEQ * R_DT) / 256, 256, 0, stream>>>(ssm_p, dtinb, L_SEQ, R_DT, NPAD);
  transpose_cvt<<<dim3(D_DIM / 32, R_DT / 32), tb, 0, stream>>>(W_dt, wdtT, R_DT, D_DIM, D_DIM);
  gemm_bf16<<<dim3(D_DIM / 64, L_SEQ / 128), 256, 0, stream>>>(dtinb, wdtT, dtb, L_SEQ, D_DIM, R_DT);
  dt_softplus<<<(L_SEQ * D_DIM) / 256, 256, 0, stream>>>(dtb, b_dt);

  // 6) fused selective scan + skip + gate -> y (bf16)
  scan_kernel<<<D_DIM / 16, 256, 0, stream>>>(dtb, hsf, ssm_p, A_log, D_par, proj, yb);

  // 7) out[L,H] = y @ W_out         (M=2048, N=2048, K=4096)
  transpose_cvt<<<dim3(H_DIM / 32, D_DIM / 32), tb, 0, stream>>>(W_out, woutT, D_DIM, H_DIM, H_DIM);
  gemm_bf16<<<dim3(H_DIM / 64, L_SEQ / 128), 256, 0, stream>>>(yb, woutT, out, L_SEQ, H_DIM, D_DIM);
}
